// Transformer_66477503807999
// MI455X (gfx1250) — matrix operand order repaired and checked by reading, not compiled, not run on hardware
//
#include <hip/hip_runtime.h>
#include <math.h>
#include <stdint.h>

#define LNUM   4
#define HNUM   16
#define DMODEL 1024
#define FDIM   4096
#define VOUT   32000
#define SEQ    2048
#define BATCH  2
#define HDIM   64
#define BS     (BATCH * SEQ)   /* 4096 rows */

#define TM 256   /* block tile M: 8 waves x 32 rows */
#define TN 64    /* block tile N */
#define TKT 64   /* K tile staged per TDM transfer */

typedef __bf16 bf16_t;
typedef __attribute__((ext_vector_type(16))) __bf16 v16bf;
typedef __attribute__((ext_vector_type(8)))  __bf16 v8bf;
typedef __attribute__((ext_vector_type(8)))  float  v8f;
typedef __attribute__((ext_vector_type(4)))  unsigned u32x4;
typedef __attribute__((ext_vector_type(8)))  int      i32x8;
typedef __attribute__((ext_vector_type(4)))  int      i32x4;

#if defined(__AMDGCN__) && __has_builtin(__builtin_amdgcn_tensor_load_to_lds) && \
    __has_builtin(__builtin_amdgcn_s_wait_tensorcnt)
#define USE_TDM 1
#else
#define USE_TDM 0
#endif

// ---------- helpers ----------

__device__ inline bf16_t f2bf(float f) {
  unsigned u = __builtin_bit_cast(unsigned, f);
  unsigned r = (u + 0x7FFFu + ((u >> 16) & 1u)) >> 16;   // RNE
  unsigned short s = (unsigned short)r;
  return __builtin_bit_cast(bf16_t, s);
}

__device__ inline v16bf cat16(v8bf a, v8bf b) {
  return __builtin_shufflevector(a, b, 0,1,2,3,4,5,6,7,8,9,10,11,12,13,14,15);
}

__device__ inline float redmax16(float v) {
  #pragma unroll
  for (int m = 1; m < 16; m <<= 1) v = fmaxf(v, __shfl_xor(v, m, 32));
  return v;
}
__device__ inline float redsum16(float v) {
  #pragma unroll
  for (int m = 1; m < 16; m <<= 1) v += __shfl_xor(v, m, 32);
  return v;
}

#if USE_TDM
// TDM: DMA a [nrows x tk] bf16 tile (row stride = K elements) into LDS.
// D# per CDNA5 ISA.
__device__ inline void tdm_load_tile(const bf16_t* gsrc, void* lds_dst,
                                     int K, int nrows, int tk) {
  unsigned long long ga = (unsigned long long)(uintptr_t)gsrc;
  unsigned lds_addr = (unsigned)(uintptr_t)lds_dst;   // generic addr low 32 = LDS offset
  u32x4 g0;
  g0[0] = 1u;                                          // count=1, no gather
  g0[1] = lds_addr;
  g0[2] = (unsigned)(ga & 0xFFFFFFFFu);
  g0[3] = ((unsigned)(ga >> 32) & 0x01FFFFFFu) | 0x80000000u;  // addr[56:32] | type=2
  int td0 = K, td1 = nrows;
  i32x8 g1;
  g1[0] = 0x00010000;                                  // data_size = 2 bytes
  g1[1] = (td0 & 0xFFFF) << 16;                        // tensor_dim0 lo16
  g1[2] = ((td0 >> 16) & 0xFFFF) | ((td1 & 0xFFFF) << 16);
  g1[3] = ((td1 >> 16) & 0xFFFF) | ((tk & 0xFFFF) << 16);   // tile_dim0 = tk
  g1[4] = nrows & 0xFFFF;                              // tile_dim1
  g1[5] = K;                                           // tensor_dim0_stride lo32
  g1[6] = 0;
  g1[7] = 0;
  i32x4 g2 = {0, 0, 0, 0};
  i32x4 g3 = {0, 0, 0, 0};
#if __clang_major__ >= 23
  i32x8 g4 = {0, 0, 0, 0, 0, 0, 0, 0};
  __builtin_amdgcn_tensor_load_to_lds(g0, g1, g2, g3, g4, 0);
#else
  __builtin_amdgcn_tensor_load_to_lds(g0, g1, g2, g3, 0);
#endif
}
#endif

// ---------- kernels ----------

// fp32 -> bf16 bulk convert (8 elems/thread; n multiple of 2048)
__global__ void cvt_bf16_k(const float* __restrict__ src, bf16_t* __restrict__ dst) {
  size_t i = ((size_t)blockIdx.x * 256 + threadIdx.x) * 8;
  float4 a = *(const float4*)(src + i);
  float4 b = *(const float4*)(src + i + 4);
  v8bf o;
  o[0] = f2bf(a.x); o[1] = f2bf(a.y); o[2] = f2bf(a.z); o[3] = f2bf(a.w);
  o[4] = f2bf(b.x); o[5] = f2bf(b.y); o[6] = f2bf(b.z); o[7] = f2bf(b.w);
  *(v8bf*)(dst + i) = o;
}

// x[b,s,d] = tok_emb[idx[b,s], d] + pe(s,d);  also bf16 copy
__global__ void embed_k(const int* __restrict__ idx, const float* __restrict__ emb,
                        float* __restrict__ x, bf16_t* __restrict__ xb) {
  size_t i = (size_t)blockIdx.x * 256 + threadIdx.x;
  int d = (int)(i % DMODEL);
  size_t bs = i / DMODEL;
  int s = (int)(bs % SEQ);
  int tok = idx[bs];
  int de = d & ~1;
  float dt = expf(-(float)de * (9.210340371976184f / (float)DMODEL));
  float ang = (float)s * dt;
  float pe = (d & 1) ? cosf(ang) : sinf(ang);
  float v = emb[(size_t)tok * DMODEL + d] + pe;
  x[i] = v;
  xb[i] = f2bf(v);
}

// Y = LN(X)*w+b ; optional bf16 copy Yb
__global__ void layernorm_k(const float* __restrict__ X, const float* __restrict__ w,
                            const float* __restrict__ b, float* __restrict__ Y,
                            bf16_t* __restrict__ Yb) {
  __shared__ float s1[256], s2[256];
  int row = blockIdx.x;
  const float* x = X + (size_t)row * DMODEL;
  float sum = 0.f, sq = 0.f;
  for (int i = threadIdx.x; i < DMODEL; i += 256) { float v = x[i]; sum += v; sq += v * v; }
  s1[threadIdx.x] = sum; s2[threadIdx.x] = sq;
  __syncthreads();
  for (int st = 128; st > 0; st >>= 1) {
    if ((int)threadIdx.x < st) { s1[threadIdx.x] += s1[threadIdx.x + st]; s2[threadIdx.x] += s2[threadIdx.x + st]; }
    __syncthreads();
  }
  float mean = s1[0] * (1.0f / DMODEL);
  float var  = s2[0] * (1.0f / DMODEL) - mean * mean;
  float inv  = rsqrtf(var + 1e-5f);
  for (int i = threadIdx.x; i < DMODEL; i += 256) {
    float v = (x[i] - mean) * inv * w[i] + b[i];
    Y[(size_t)row * DMODEL + i] = v;
    if (Yb) Yb[(size_t)row * DMODEL + i] = f2bf(v);
  }
}

// C[M,N] = A[M,K](bf16) @ W[N,K]^T(bf16) (+bias)(+res)(gelu); outputs fp32 Cf and/or bf16 Cb.
// 256 threads = 8 waves; block tile 256x64; each wave computes 32x64 (8 accumulators).
// B tile (64x64 bf16, 8KB) staged in LDS by the Tensor Data Mover, double-buffered so
// the DMA of tile t+1 overlaps the WMMAs on tile t (wait tensorcnt<=1; TDM is in-order).
__global__ __launch_bounds__(256) void gemm_wmma(
    const bf16_t* __restrict__ A, const bf16_t* __restrict__ W,
    const float* __restrict__ bias, const float* __restrict__ res,
    float* __restrict__ Cf, bf16_t* __restrict__ Cb,
    int M, int N, int K, int gelu_flag) {
  __shared__ __attribute__((aligned(16))) bf16_t bsh[2][TN * TKT];   // 2 x 8KB
  int tid  = threadIdx.x;
  int lane = tid & 31, wave = tid >> 5;
  int lm = lane & 15, hi = lane >> 4;
  int n0 = blockIdx.x * TN;
  int m0 = blockIdx.y * TM + wave * 32;
  const bf16_t* arow0 = A + (size_t)(m0 + lm) * K;
  const bf16_t* arow1 = arow0 + (size_t)16 * K;
  int koffA = hi ? 8 : 0;    // A frag: lanes 0-15 K{0..7,16..23}; 16-31 K{8..15,24..31}
  int koffB = hi ? 8 : 0;    // B frag: lanes 0-15 K{0..7,16..23}; 16-31 K{8..15,24..31}

  v8f acc[2][4];
  #pragma unroll
  for (int mt = 0; mt < 2; ++mt)
    #pragma unroll
    for (int nt = 0; nt < 4; ++nt)
      acc[mt][nt] = (v8f){0.f,0.f,0.f,0.f,0.f,0.f,0.f,0.f};

  const int ntile = K / TKT;
#if USE_TDM
  if (wave == 0) tdm_load_tile(W + (size_t)n0 * K, &bsh[0][0], K, TN, TKT);
#endif
  for (int t = 0; t < ntile; ++t) {
    const int k0 = t * TKT;
    bf16_t* buf = &bsh[t & 1][0];
#if USE_TDM
    if (wave == 0) {
      if (t + 1 < ntile) {
        tdm_load_tile(W + (size_t)n0 * K + (size_t)(t + 1) * TKT,
                      &bsh[(t + 1) & 1][0], K, TN, TKT);
        __builtin_amdgcn_s_wait_tensorcnt(1);   // oldest (current tile) done
      } else {
        __builtin_amdgcn_s_wait_tensorcnt(0);   // drain on last tile
      }
    }
#else
    #pragma unroll
    for (int c = 0; c < 2; ++c) {
      int ch = tid * 2 + c;                  // 0..511 chunks of 8 elems
      int row = ch >> 3, off = (ch & 7) * 8;
      *(v8bf*)&buf[row * TKT + off] =
          *(const v8bf*)(W + (size_t)(n0 + row) * K + k0 + off);
    }
#endif
    __syncthreads();

    if (k0 + TKT < K) {
      __builtin_prefetch(arow0 + k0 + TKT, 0, 1);
      __builtin_prefetch(arow1 + k0 + TKT, 0, 1);
    }

    #pragma unroll
    for (int kk = 0; kk < TKT; kk += 32) {
      v16bf af0 = cat16(*(const v8bf*)(arow0 + k0 + kk + koffA),
                        *(const v8bf*)(arow0 + k0 + kk + 16 + koffA));
      v16bf af1 = cat16(*(const v8bf*)(arow1 + k0 + kk + koffA),
                        *(const v8bf*)(arow1 + k0 + kk + 16 + koffA));
      #pragma unroll
      for (int nt = 0; nt < 4; ++nt) {
        const bf16_t* bp = &buf[(nt * 16 + lm) * TKT + kk + koffB];
        v16bf bfr = cat16(*(const v8bf*)bp, *(const v8bf*)(bp + 16));
        acc[0][nt] = __builtin_amdgcn_wmma_f32_16x16x32_bf16(false, af0, false, bfr, (short)0,
                                                             acc[0][nt], false, false);
        acc[1][nt] = __builtin_amdgcn_wmma_f32_16x16x32_bf16(false, af1, false, bfr, (short)0,
                                                             acc[1][nt], false, false);
      }
    }
    __syncthreads();
  }

  #pragma unroll
  for (int mt = 0; mt < 2; ++mt) {
    #pragma unroll
    for (int nt = 0; nt < 4; ++nt) {
      int col = n0 + nt * 16 + lm;
      float bv = bias ? bias[col] : 0.f;
      #pragma unroll
      for (int r = 0; r < 8; ++r) {
        int row = m0 + mt * 16 + r + hi * 8;
        float v = acc[mt][nt][r] + bv;
        if (res) v += res[(size_t)row * N + col];
        if (gelu_flag) v = 0.5f * v * (1.0f + erff(v * 0.7071067811865475f));
        if (Cf) Cf[(size_t)row * N + col] = v;
        if (Cb) Cb[(size_t)row * N + col] = f2bf(v);
      }
    }
  }
}

// Flash attention on bf16 qkv [B,S,3D]; writes bf16 attn-out [B,S,D].
__global__ __launch_bounds__(32) void flash_attn_k(const bf16_t* __restrict__ qkv,
                                                   bf16_t* __restrict__ out) {
  __shared__ bf16_t pl[16 * 32];
  int lane = threadIdx.x & 31;
  int lm = lane & 15, hi = lane >> 4;
  int row0 = blockIdx.x * 16;
  int b = blockIdx.y / HNUM, h = blockIdx.y % HNUM;
  const size_t R3D = (size_t)3 * DMODEL;
  const bf16_t* base = qkv + (size_t)b * SEQ * R3D + (size_t)h * HDIM;
  int koffA = hi ? 16 : 0;
  int koffB = hi ? 16 : 0;
  const float scale = 0.125f;  // 1/sqrt(64)

  const bf16_t* qrow = base + (size_t)(row0 + lm) * R3D;
  v16bf qa0 = cat16(*(const v8bf*)(qrow + koffA),      *(const v8bf*)(qrow + 8 + koffA));
  v16bf qa1 = cat16(*(const v8bf*)(qrow + 32 + koffA), *(const v8bf*)(qrow + 40 + koffA));

  v8f o[4];
  #pragma unroll
  for (int c = 0; c < 4; ++c) o[c] = (v8f){0.f,0.f,0.f,0.f,0.f,0.f,0.f,0.f};
  float rmax[8], rsum[8];
  #pragma unroll
  for (int r = 0; r < 8; ++r) { rmax[r] = -INFINITY; rsum[r] = 0.f; }

  for (int j0 = 0; j0 <= row0 + 15; j0 += 32) {
    float sv[16];
    #pragma unroll
    for (int c = 0; c < 2; ++c) {
      const bf16_t* krow = base + DMODEL + (size_t)(j0 + c * 16 + lm) * R3D;
      v16bf kb0 = cat16(*(const v8bf*)(krow + koffB),      *(const v8bf*)(krow + koffB + 8));
      v16bf kb1 = cat16(*(const v8bf*)(krow + 32 + koffB), *(const v8bf*)(krow + 40 + koffB));
      v8f a = {0.f,0.f,0.f,0.f,0.f,0.f,0.f,0.f};
      a = __builtin_amdgcn_wmma_f32_16x16x32_bf16(false, qa0, false, kb0, (short)0, a, false, false);
      a = __builtin_amdgcn_wmma_f32_16x16x32_bf16(false, qa1, false, kb1, (short)0, a, false, false);
      int jcol = j0 + c * 16 + lm;
      #pragma unroll
      for (int r = 0; r < 8; ++r) {
        int row = row0 + r + hi * 8;
        sv[c * 8 + r] = (jcol > row) ? -INFINITY : a[r] * scale;
      }
    }
    #pragma unroll
    for (int r = 0; r < 8; ++r) {
      float tm = redmax16(fmaxf(sv[r], sv[8 + r]));
      float nm = fmaxf(rmax[r], tm);
      float corr = expf(rmax[r] - nm);
      float p0 = expf(sv[r] - nm);
      float p1 = expf(sv[8 + r] - nm);
      sv[r] = p0; sv[8 + r] = p1;
      float ps = redsum16(p0 + p1);
      rsum[r] = rsum[r] * corr + ps;
      rmax[r] = nm;
      o[0][r] *= corr; o[1][r] *= corr; o[2][r] *= corr; o[3][r] *= corr;
    }
    __syncthreads();
    #pragma unroll
    for (int c = 0; c < 2; ++c)
      #pragma unroll
      for (int r = 0; r < 8; ++r)
        pl[(r + hi * 8) * 32 + c * 16 + lm] = f2bf(sv[c * 8 + r]);
    __syncthreads();
    v16bf pa;
    #pragma unroll
    for (int i = 0; i < 8; ++i) {
      pa[i]     = pl[lm * 32 + koffA + i];
      pa[8 + i] = pl[lm * 32 + 8 + koffA + i];
    }
    const bf16_t* vbase = qkv + ((size_t)b * SEQ + j0) * R3D + 2 * DMODEL + (size_t)h * HDIM;
    #pragma unroll
    for (int c2 = 0; c2 < 4; ++c2) {
      int d = c2 * 16 + lm;
      v16bf vb;
      #pragma unroll
      for (int i = 0; i < 16; ++i)
        vb[i] = vbase[(size_t)(koffB + i) * R3D + d];
      o[c2] = __builtin_amdgcn_wmma_f32_16x16x32_bf16(false, pa, false, vb, (short)0, o[c2],
                                                      false, false);
    }
    __syncthreads();
  }

  #pragma unroll
  for (int c2 = 0; c2 < 4; ++c2) {
    int d = c2 * 16 + lm;
    #pragma unroll
    for (int r = 0; r < 8; ++r) {
      int row = row0 + r + hi * 8;
      out[((size_t)b * SEQ + row) * DMODEL + h * HDIM + d] = f2bf(o[c2][r] / rsum[r]);
    }
  }
}

// ---------- host ----------

extern "C" void kernel_launch(void* const* d_in, const int* in_sizes, int n_in,
                              void* d_out, int out_size, void* d_ws, size_t ws_size,
                              hipStream_t stream) {
  const int*   idx    = (const int*)d_in[0];
  const float* tokemb = (const float*)d_in[1];
  const float* qkv_w  = (const float*)d_in[2];
  const float* o_w    = (const float*)d_in[3];
  const float* w1     = (const float*)d_in[4];
  const float* b1     = (const float*)d_in[5];
  const float* w2     = (const float*)d_in[6];
  const float* b2     = (const float*)d_in[7];
  const float* ln1w   = (const float*)d_in[8];
  const float* ln1b   = (const float*)d_in[9];
  const float* ln2w   = (const float*)d_in[10];
  const float* ln2b   = (const float*)d_in[11];
  const float* lnfw   = (const float*)d_in[12];
  const float* lnfb   = (const float*)d_in[13];
  const float* decw   = (const float*)d_in[14];
  const float* decb   = (const float*)d_in[15];
  float* out = (float*)d_out;

  // workspace carve
  char* p = (char*)d_ws;
  auto carve = [&](size_t bytes) { void* r = (void*)p; p += (bytes + 255) & ~(size_t)255; return r; };
  const size_t NX = (size_t)BS * DMODEL;
  float*  x       = (float*)carve(NX * 4);
  float*  y       = (float*)carve(NX * 4);
  bf16_t* x_bf    = (bf16_t*)carve(NX * 2);
  bf16_t* qkv_bf  = (bf16_t*)carve((size_t)BS * 3 * DMODEL * 2);
  bf16_t* attn_bf = (bf16_t*)carve(NX * 2);
  bf16_t* h_bf    = (bf16_t*)carve((size_t)BS * FDIM * 2);
  bf16_t* wq_bf   = (bf16_t*)carve((size_t)LNUM * 3 * DMODEL * DMODEL * 2);
  bf16_t* wo_bf   = (bf16_t*)carve((size_t)LNUM * DMODEL * DMODEL * 2);
  bf16_t* w1_bf   = (bf16_t*)carve((size_t)LNUM * FDIM * DMODEL * 2);
  bf16_t* w2_bf   = (bf16_t*)carve((size_t)LNUM * DMODEL * FDIM * 2);
  bf16_t* dec_bf  = (bf16_t*)carve((size_t)VOUT * DMODEL * 2);

  // convert all weights to bf16 (counts are multiples of 2048)
  cvt_bf16_k<<<(LNUM * 3 * DMODEL * DMODEL) / 2048, 256, 0, stream>>>(qkv_w, wq_bf);
  cvt_bf16_k<<<(LNUM * DMODEL * DMODEL) / 2048, 256, 0, stream>>>(o_w, wo_bf);
  cvt_bf16_k<<<(LNUM * FDIM * DMODEL) / 2048, 256, 0, stream>>>(w1, w1_bf);
  cvt_bf16_k<<<(LNUM * DMODEL * FDIM) / 2048, 256, 0, stream>>>(w2, w2_bf);
  cvt_bf16_k<<<(VOUT * DMODEL) / 2048, 256, 0, stream>>>(decw, dec_bf);

  embed_k<<<(BS * DMODEL) / 256, 256, 0, stream>>>(idx, tokemb, x, x_bf);

  for (int l = 0; l < LNUM; ++l) {
    const bf16_t* qw  = wq_bf + (size_t)l * 3 * DMODEL * DMODEL;
    const bf16_t* ow  = wo_bf + (size_t)l * DMODEL * DMODEL;
    const bf16_t* w1l = w1_bf + (size_t)l * FDIM * DMODEL;
    const bf16_t* w2l = w2_bf + (size_t)l * DMODEL * FDIM;

    // qkv(bf16) = x @ qkv_w^T
    gemm_wmma<<<dim3(3 * DMODEL / TN, BS / TM), 256, 0, stream>>>(
        x_bf, qw, nullptr, nullptr, nullptr, qkv_bf, BS, 3 * DMODEL, DMODEL, 0);
    // attention -> attn_bf
    flash_attn_k<<<dim3(SEQ / 16, BATCH * HNUM), 32, 0, stream>>>(qkv_bf, attn_bf);
    // y = x + attn @ o_w^T ; x = LN1(y) (fp32 + bf16)
    gemm_wmma<<<dim3(DMODEL / TN, BS / TM), 256, 0, stream>>>(
        attn_bf, ow, nullptr, x, y, nullptr, BS, DMODEL, DMODEL, 0);
    layernorm_k<<<BS, 256, 0, stream>>>(y, ln1w + l * DMODEL, ln1b + l * DMODEL, x, x_bf);
    // h(bf16) = gelu(x @ w1^T + b1)
    gemm_wmma<<<dim3(FDIM / TN, BS / TM), 256, 0, stream>>>(
        x_bf, w1l, b1 + (size_t)l * FDIM, nullptr, nullptr, h_bf, BS, FDIM, DMODEL, 1);
    // y = x + h @ w2^T + b2 ; x = LN2(y)
    gemm_wmma<<<dim3(DMODEL / TN, BS / TM), 256, 0, stream>>>(
        h_bf, w2l, b2 + (size_t)l * DMODEL, x, y, nullptr, BS, DMODEL, FDIM, 0);
    layernorm_k<<<BS, 256, 0, stream>>>(y, ln2w + l * DMODEL, ln2b + l * DMODEL, x, x_bf);
  }

  // final LN -> x_bf ; logits = xf @ dec_w^T + dec_b
  layernorm_k<<<BS, 256, 0, stream>>>(x, lnfw, lnfb, y, x_bf);
  gemm_wmma<<<dim3(VOUT / TN, BS / TM), 256, 0, stream>>>(
      x_bf, dec_bf, decb, nullptr, out, nullptr, BS, VOUT, DMODEL, 0);
}
